// TDNN_72490458022265
// MI455X (gfx1250) — matrix operand order repaired and checked by reading, not compiled, not run on hardware
//
#include <hip/hip_runtime.h>
#include <math.h>

typedef __attribute__((ext_vector_type(8)))  __bf16 v8bf;
typedef __attribute__((ext_vector_type(16))) __bf16 v16bf;
typedef __attribute__((ext_vector_type(8)))  float  v8f;

#define BATCH 16
#define TPAD  2048
#define CH    512
#define NSPKR 6000
#define C4PAD 1504
#define C4    1500
#define TOUT4 2034

__device__ __forceinline__ __bf16 f2bf(float f) { return (__bf16)f; }

// B fragment: lane holds K {base..base+7} and {base+16..base+23} (lane = output column, K runs along row of W)
__device__ __forceinline__ v16bf ldb16(const __bf16* p) {
  v8bf lo = *(const v8bf*)p;
  v8bf hi = *(const v8bf*)(p + 16);
  return __builtin_shufflevector(lo, hi, 0,1,2,3,4,5,6,7,8,9,10,11,12,13,14,15);
}
// A fragment: lane holds K {base..base+7} and {base+16..base+23}
__device__ __forceinline__ v16bf lda16(const __bf16* p) {
  v8bf lo = *(const v8bf*)p;
  v8bf hi = *(const v8bf*)(p + 16);
  return __builtin_shufflevector(lo, hi, 0,1,2,3,4,5,6,7,8,9,10,11,12,13,14,15);
}
__device__ __forceinline__ v8f wmma_bf16(v16bf a, v16bf b, v8f c) {
  return __builtin_amdgcn_wmma_f32_16x16x32_bf16(false, a, false, b, (short)0, c, false, false);
}

// ---------------- weight conversion ----------------
__global__ void k_w2bf(const float* __restrict__ src, __bf16* __restrict__ dst, int n) {
  int i = blockIdx.x * blockDim.x + threadIdx.x;
  if (i < n) dst[i] = f2bf(src[i]);
}
// tdnn4 weights padded 1500 -> 1504 rows (zeros)
__global__ void k_w4bf(const float* __restrict__ src, __bf16* __restrict__ dst) {
  int i = blockIdx.x * blockDim.x + threadIdx.x;
  if (i >= C4PAD * CH) return;
  int r = i >> 9, c = i & 511;
  dst[i] = f2bf(r < C4 ? src[r * CH + c] : 0.f);
}

// ---------------- projection: [B,T,20] x [512,20]^T + b -> bf16 [B,2048,512] ----
__global__ void k_proj(const float* __restrict__ mfcc, const float* __restrict__ w,
                       const float* __restrict__ bias, __bf16* __restrict__ X0) {
  int idx = blockIdx.x * blockDim.x + threadIdx.x;   // B*T*512 threads
  int c = idx & 511;
  int bt = idx >> 9;
  const float* mr = mfcc + (size_t)bt * 20;
  const float* wr = w + (size_t)c * 20;
  float a = bias[c];
#pragma unroll
  for (int f = 0; f < 20; ++f) a = fmaf(mr[f], wr[f], a);
  X0[idx] = f2bf(a);      // no ReLU on proj
}

// ---------------- TDNN conv layer (implicit im2col GEMM via WMMA) ----------------
// wave tile: 32 time x 64 out channels (2 time tiles x 4 och tiles; B frags reused 2x,
// A frags reused 4x). Distinct fragment variables -> scheduler can batch loads and
// avoid a full loadcnt drain before every WMMA.
__global__ void __launch_bounds__(32)
k_tdnn(const __bf16* __restrict__ X, const __bf16* __restrict__ W,
       const float* __restrict__ bias, __bf16* __restrict__ Y,
       int Tin, int Tout, int taps, int dil) {
  const int lane   = threadIdx.x;
  const int laneM  = lane & 15;
  const int laneHi = lane >> 4;
  const int t0 = blockIdx.x * 32;
  const int o0 = blockIdx.y * 64;
  const int b  = blockIdx.z;
  const int Ktot = taps * CH;
  const __bf16* Xb = X + (size_t)b * TPAD * CH;

  v8f zero = {};
  v8f acc0[4] = {zero, zero, zero, zero};   // time tile 0
  v8f acc1[4] = {zero, zero, zero, zero};   // time tile 1

  for (int j = 0; j < taps; ++j) {
    int r0 = t0 + laneM + j * dil;        if (r0 > Tin - 1) r0 = Tin - 1;
    int r1 = t0 + 16 + laneM + j * dil;   if (r1 > Tin - 1) r1 = Tin - 1;
    const __bf16* x0 = Xb + (size_t)r0 * CH + laneHi * 8;
    const __bf16* x1 = Xb + (size_t)r1 * CH + laneHi * 8;
#pragma unroll 4
    for (int cc = 0; cc < CH; cc += 32) {
      v16bf a0 = lda16(x0 + cc);
      v16bf a1 = lda16(x1 + cc);
      const int kc = j * CH + cc + laneHi * 8;
      const __bf16* wp = W + (size_t)(o0 + laneM) * Ktot + kc;
      v16bf b0 = ldb16(wp);
      v16bf b1 = ldb16(wp + (size_t)16 * Ktot);
      v16bf b2 = ldb16(wp + (size_t)32 * Ktot);
      v16bf b3 = ldb16(wp + (size_t)48 * Ktot);
      acc0[0] = wmma_bf16(a0, b0, acc0[0]);
      acc1[0] = wmma_bf16(a1, b0, acc1[0]);
      acc0[1] = wmma_bf16(a0, b1, acc0[1]);
      acc1[1] = wmma_bf16(a1, b1, acc1[1]);
      acc0[2] = wmma_bf16(a0, b2, acc0[2]);
      acc1[2] = wmma_bf16(a1, b2, acc1[2]);
      acc0[3] = wmma_bf16(a0, b3, acc0[3]);
      acc1[3] = wmma_bf16(a1, b3, acc1[3]);
    }
  }
#pragma unroll
  for (int i = 0; i < 4; ++i) {
    const int o = o0 + i * 16 + laneM;
    const float bv = bias[o];
#pragma unroll
    for (int r = 0; r < 8; ++r) {
      const int t = t0 + r + laneHi * 8;            // D layout: M = r + 8*laneHi, N = laneM
      if (t < Tout) {
        float v = acc0[i][r] + bv;
        v = v > 0.f ? v : 0.f;
        Y[((size_t)b * TPAD + t) * CH + o] = f2bf(v);
      }
      const int t2 = t + 16;
      if (t2 < Tout) {
        float v = acc1[i][r] + bv;
        v = v > 0.f ? v : 0.f;
        Y[((size_t)b * TPAD + t2) * CH + o] = f2bf(v);
      }
    }
  }
}

// ---------------- TDNN layer 4 (512 -> 1500, k=1) fused with masked stats pooling ----
__global__ void __launch_bounds__(32)
k_tdnn4(const __bf16* __restrict__ X, const __bf16* __restrict__ W,
        const float* __restrict__ bias, const int* __restrict__ src_len,
        float* __restrict__ sumo, float* __restrict__ sqo) {
  const int lane   = threadIdx.x;
  const int laneM  = lane & 15;
  const int laneHi = lane >> 4;
  const int o0 = blockIdx.x * 64;                   // 24 groups -> 1536 >= 1504 padded chans
  const int b  = blockIdx.y;
  int len = src_len[b] - 8;
  if (len > TOUT4) len = TOUT4;
  const __bf16* Xb = X + (size_t)b * TPAD * CH;

  float s[4] = {0.f, 0.f, 0.f, 0.f}, q[4] = {0.f, 0.f, 0.f, 0.f};
  v8f zero = {};

  for (int tt = 0; tt < 64; ++tt) {                 // two 16-row tiles per iteration
    const int t0 = tt * 32;
    v8f acc0[4] = {zero, zero, zero, zero};
    v8f acc1[4] = {zero, zero, zero, zero};
    int r0 = t0 + laneM;        if (r0 > TOUT4 - 1) r0 = TOUT4 - 1;
    int r1 = t0 + 16 + laneM;   if (r1 > TOUT4 - 1) r1 = TOUT4 - 1;
    const __bf16* x0 = Xb + (size_t)r0 * CH + laneHi * 8;
    const __bf16* x1 = Xb + (size_t)r1 * CH + laneHi * 8;
#pragma unroll 4
    for (int cc = 0; cc < CH; cc += 32) {
      v16bf a0 = lda16(x0 + cc);
      v16bf a1 = lda16(x1 + cc);
      const int kc = cc + laneHi * 8;
      const __bf16* wp = W + (size_t)(o0 + laneM) * CH + kc;
#pragma unroll
      for (int i = 0; i < 4; ++i) {
        if (o0 + i * 16 < C4PAD) {                  // uniform per wave
          v16bf bf = ldb16(wp + (size_t)(i * 16) * CH);
          acc0[i] = wmma_bf16(a0, bf, acc0[i]);
          acc1[i] = wmma_bf16(a1, bf, acc1[i]);
        }
      }
    }
#pragma unroll
    for (int i = 0; i < 4; ++i) {
      const int o = o0 + i * 16 + laneM;
      const float bv = (o < C4) ? bias[o] : 0.f;
#pragma unroll
      for (int r = 0; r < 8; ++r) {
        const int t = t0 + r + laneHi * 8;
        float v = acc0[i][r] + bv;
        v = v > 0.f ? v : 0.f;
        if (t < len) { s[i] += v; q[i] += v * v; }
        const int t2 = t + 16;
        float v2 = acc1[i][r] + bv;
        v2 = v2 > 0.f ? v2 : 0.f;
        if (t2 < len) { s[i] += v2; q[i] += v2 * v2; }
      }
    }
  }
#pragma unroll
  for (int i = 0; i < 4; ++i) {
    float ss = s[i] + __shfl_xor(s[i], 16, 32);     // combine the two time-halves per channel
    float qq = q[i] + __shfl_xor(q[i], 16, 32);
    const int o = o0 + i * 16 + laneM;
    if (laneHi == 0 && o < C4) {
      sumo[b * C4 + o] = ss;
      sqo [b * C4 + o] = qq;
    }
  }
}

// ---------------- stats finalize: mean + unbiased std -> stat [B,3000] ----------------
__global__ void k_stats(const float* __restrict__ sumo, const float* __restrict__ sqo,
                        const int* __restrict__ src_len, float* __restrict__ stat) {
  int idx = blockIdx.x * blockDim.x + threadIdx.x;
  if (idx >= BATCH * C4) return;
  int b = idx / C4, c = idx - b * C4;
  int li = src_len[b] - 8;
  if (li > TOUT4) li = TOUT4;
  float n = (float)li;
  float mean = sumo[idx] / n;
  float var = (sqo[idx] - n * mean * mean) / (n - 1.f);
  stat[b * 3000 + c] = mean;
  stat[b * 3000 + C4 + c] = sqrtf(fmaxf(var, 0.f));
}

// ---------------- fe: xvec = stat @ fe_w^T + fe_b (also writes xvec output slot) ------
__global__ void k_fe(const float* __restrict__ stat, const float* __restrict__ w,
                     const float* __restrict__ bias, float* __restrict__ xvec,
                     float* __restrict__ out_x) {
  int idx = blockIdx.x * blockDim.x + threadIdx.x;
  if (idx >= BATCH * 512) return;
  int b = idx >> 9, o = idx & 511;
  const float* sr = stat + (size_t)b * 3000;
  const float* wr = w + (size_t)o * 3000;
  float a = bias[o];
  for (int k = 0; k < 3000; ++k) a = fmaf(sr[k], wr[k], a);
  xvec[idx] = a;
  out_x[idx] = a;
}

// ---------------- generic 512-in dense: y = x @ w^T + b ----------------
__global__ void k_dense512(const float* __restrict__ x, const float* __restrict__ w,
                           const float* __restrict__ bias, float* __restrict__ y, int outd) {
  int idx = blockIdx.x * blockDim.x + threadIdx.x;
  if (idx >= BATCH * outd) return;
  int b = idx / outd, o = idx - b * outd;
  const float* xr = x + (size_t)b * 512;
  const float* wr = w + (size_t)o * 512;
  float a = bias[o];
  for (int k = 0; k < 512; ++k) a = fmaf(xr[k], wr[k], a);
  y[idx] = a;
}

__global__ void k_rownorm(const float* __restrict__ hs, float* __restrict__ hsn) {
  int b = threadIdx.x;
  if (b >= BATCH) return;
  float a = 0.f;
  for (int k = 0; k < 512; ++k) { float v = hs[b * 512 + k]; a = fmaf(v, v, a); }
  hsn[b] = sqrtf(a);
}

__global__ void k_colnorm(const float* __restrict__ spk_w, float* __restrict__ cn) {
  int s = blockIdx.x * blockDim.x + threadIdx.x;
  if (s >= NSPKR) return;
  float a = 0.f;
  for (int k = 0; k < 512; ++k) { float v = spk_w[(size_t)k * NSPKR + s]; a = fmaf(v, v, a); }
  cn[s] = sqrtf(a);
}

__global__ void k_cos(const float* __restrict__ hs, const float* __restrict__ spk_w,
                      const float* __restrict__ hsn, const float* __restrict__ cn,
                      float* __restrict__ cosb) {
  int idx = blockIdx.x * blockDim.x + threadIdx.x;
  if (idx >= BATCH * NSPKR) return;
  int b = idx / NSPKR, s = idx - b * NSPKR;
  const float* hr = hs + (size_t)b * 512;
  float a = 0.f;
  for (int k = 0; k < 512; ++k) a = fmaf(hr[k], spk_w[(size_t)k * NSPKR + s], a);
  cosb[idx] = a / (hsn[b] * cn[s]);
}

__global__ void k_topk(const float* __restrict__ cosb, int* __restrict__ top) {
  int b = threadIdx.x;
  if (b >= BATCH) return;
  const float* row = cosb + (size_t)b * NSPKR;
  int sel[6];
  for (int k = 0; k < 6; ++k) {
    float best = -3.4e38f; int bi = 0;
    for (int s = 0; s < NSPKR; ++s) {
      bool used = false;
      for (int j = 0; j < k; ++j) used |= (sel[j] == s);
      float v = row[s];
      if (!used && v > best) { best = v; bi = s; }
    }
    sel[k] = bi;
    top[b * 6 + k] = bi;
  }
}

__global__ void k_logits(const float* __restrict__ cosb, const int* __restrict__ spkers,
                         const int* __restrict__ top, float* __restrict__ out) {
  int idx = blockIdx.x * blockDim.x + threadIdx.x;
  if (idx >= BATCH * NSPKR) return;
  int b = idx / NSPKR, s = idx - b * NSPKR;
  float c = cosb[idx];
  bool pen = false;
#pragma unroll
  for (int j = 0; j < 6; ++j) pen |= (top[b * 6 + j] == s);
  float v = (spkers[b] == s) ? (c - 0.4f) : (pen ? (c + 0.1f) : c);
  out[idx] = 30.f * v;
}

__global__ void k_lang1(const float* __restrict__ lhid, const float* __restrict__ w,
                        const float* __restrict__ bias, float* __restrict__ out) {
  int idx = threadIdx.x;
  if (idx >= BATCH * 2) return;
  int b = idx >> 1, r = idx & 1;
  const float* hr = lhid + (size_t)b * 512;
  const float* wr = w + (size_t)r * 512;
  float a = bias[r];
  for (int k = 0; k < 512; ++k) a = fmaf(hr[k], wr[k], a);
  out[idx] = a;
}

extern "C" void kernel_launch(void* const* d_in, const int* in_sizes, int n_in,
                              void* d_out, int out_size, void* d_ws, size_t ws_size,
                              hipStream_t stream) {
  (void)in_sizes; (void)n_in; (void)out_size; (void)ws_size;
  const float* mfcc    = (const float*)d_in[0];
  const int*   src_len = (const int*)  d_in[1];
  const int*   spkers  = (const int*)  d_in[2];
  const float* proj_w  = (const float*)d_in[3];
  const float* proj_b  = (const float*)d_in[4];
  const float* tw[5] = {(const float*)d_in[5], (const float*)d_in[7], (const float*)d_in[9],
                        (const float*)d_in[11], (const float*)d_in[13]};
  const float* tb[5] = {(const float*)d_in[6], (const float*)d_in[8], (const float*)d_in[10],
                        (const float*)d_in[12], (const float*)d_in[14]};
  const float* fe_w  = (const float*)d_in[15];
  const float* fe_b  = (const float*)d_in[16];
  const float* cls_w = (const float*)d_in[17];
  const float* cls_b = (const float*)d_in[18];
  const float* spk_w = (const float*)d_in[19];
  const float* l0w   = (const float*)d_in[20];
  const float* l0b   = (const float*)d_in[21];
  const float* l1w   = (const float*)d_in[22];
  const float* l1b   = (const float*)d_in[23];
  float* out = (float*)d_out;

  char* ws = (char*)d_ws;
  size_t off = 0;
  auto alloc = [&](size_t bytes) -> void* {
    void* p = ws + off;
    off += (bytes + 255) & ~(size_t)255;
    return p;
  };
  __bf16* X0 = (__bf16*)alloc((size_t)BATCH * TPAD * CH * 2);
  __bf16* X1 = (__bf16*)alloc((size_t)BATCH * TPAD * CH * 2);
  __bf16* W0 = (__bf16*)alloc((size_t)512 * 2560 * 2);
  __bf16* W1 = (__bf16*)alloc((size_t)512 * 1536 * 2);
  __bf16* W2 = (__bf16*)alloc((size_t)512 * 1536 * 2);
  __bf16* W3 = (__bf16*)alloc((size_t)512 * 512 * 2);
  __bf16* W4 = (__bf16*)alloc((size_t)C4PAD * 512 * 2);
  float* sumo = (float*)alloc((size_t)BATCH * C4 * 4);
  float* sqo  = (float*)alloc((size_t)BATCH * C4 * 4);
  float* stat = (float*)alloc((size_t)BATCH * 3000 * 4);
  float* xvec = (float*)alloc((size_t)BATCH * 512 * 4);
  float* hs   = (float*)alloc((size_t)BATCH * 512 * 4);
  float* hsn  = (float*)alloc(64);
  float* cn   = (float*)alloc((size_t)NSPKR * 4);
  float* cosb = (float*)alloc((size_t)BATCH * NSPKR * 4);
  int*   top  = (int*)alloc((size_t)BATCH * 6 * 4);
  float* lhid = (float*)alloc((size_t)BATCH * 512 * 4);

  // weight conversion to bf16
  k_w2bf<<<(512 * 2560 + 255) / 256, 256, 0, stream>>>(tw[0], W0, 512 * 2560);
  k_w2bf<<<(512 * 1536 + 255) / 256, 256, 0, stream>>>(tw[1], W1, 512 * 1536);
  k_w2bf<<<(512 * 1536 + 255) / 256, 256, 0, stream>>>(tw[2], W2, 512 * 1536);
  k_w2bf<<<(512 * 512  + 255) / 256, 256, 0, stream>>>(tw[3], W3, 512 * 512);
  k_w4bf<<<(C4PAD * 512 + 255) / 256, 256, 0, stream>>>(tw[4], W4);

  // projection
  k_proj<<<(BATCH * TPAD * CH) / 256, 256, 0, stream>>>(mfcc, proj_w, proj_b, X0);

  // TDNN stack (ping-pong X0/X1), WMMA bf16; grid.x covers 32 time rows per block
  dim3 gt(64, 8, BATCH), blk(32);
  k_tdnn<<<gt, blk, 0, stream>>>(X0, W0, tb[0], X1, 2048, 2044, 5, 1);
  k_tdnn<<<gt, blk, 0, stream>>>(X1, W1, tb[1], X0, 2044, 2040, 3, 2);
  k_tdnn<<<gt, blk, 0, stream>>>(X0, W2, tb[2], X1, 2040, 2034, 3, 3);
  k_tdnn<<<gt, blk, 0, stream>>>(X1, W3, tb[3], X0, 2034, 2034, 1, 1);
  k_tdnn4<<<dim3(24, BATCH), blk, 0, stream>>>(X0, W4, tb[4], src_len, sumo, sqo);

  // pooled stats -> embedding -> heads
  k_stats<<<(BATCH * C4 + 255) / 256, 256, 0, stream>>>(sumo, sqo, src_len, stat);
  k_fe<<<(BATCH * 512 + 255) / 256, 256, 0, stream>>>(stat, fe_w, fe_b, xvec, out + 96032);
  k_dense512<<<(BATCH * 512 + 255) / 256, 256, 0, stream>>>(xvec, cls_w, cls_b, hs, 512);
  k_rownorm<<<1, 16, 0, stream>>>(hs, hsn);
  k_colnorm<<<(NSPKR + 255) / 256, 256, 0, stream>>>(spk_w, cn);
  k_cos<<<(BATCH * NSPKR + 255) / 256, 256, 0, stream>>>(hs, spk_w, hsn, cn, cosb);
  k_topk<<<1, 16, 0, stream>>>(cosb, top);
  k_logits<<<(BATCH * NSPKR + 255) / 256, 256, 0, stream>>>(cosb, spkers, top, out);
  k_dense512<<<(BATCH * 512 + 255) / 256, 256, 0, stream>>>(xvec, l0w, l0b, lhid, 512);
  k_lang1<<<1, 32, 0, stream>>>(lhid, l1w, l1b, out + 96000);
}
